// MultiScaleRetention_32186484916893
// MI455X (gfx1250) — matrix operand order repaired and checked by reading, not compiled, not run on hardware
//
#include <hip/hip_runtime.h>
#include <hip/hip_bf16.h>
#include <math.h>
#include <stdint.h>

typedef __attribute__((ext_vector_type(16))) __bf16      v16bf;
typedef __attribute__((ext_vector_type(8)))  float       v8f;
typedef __attribute__((ext_vector_type(4)))  unsigned int v4u;

constexpr int B  = 2;
constexpr int S  = 2048;
constexpr int E  = 1024;
constexpr int VE = 2048;
constexpr int H  = 8;
constexpr int DK = E / H;    // 128
constexpr int DV = VE / H;   // 256
constexpr int M  = B * S;    // 4096

union FragU { v16bf v; v4u u[2]; };

// A-fragment (16x32, 16-bit): lane in [0,15]: row=lane, K 0..15;
// lane in [16,31]: same row, K 16..31 (contiguous, matches load_fragB).
__device__ __forceinline__ v16bf load_fragA(const __bf16* base, int ld, int lane) {
  const int off = (lane & 15) * ld + ((lane >> 4) << 4);
  FragU f;
  f.u[0] = *(const v4u*)(base + off);
  f.u[1] = *(const v4u*)(base + off + 8);
  return f.v;
}

// B-fragment (32x16, 16-bit) from row-major (N x K) memory: lane gives N=lane&15,
// K = (lane>>4)*16 + 0..15 contiguous.
__device__ __forceinline__ v16bf load_fragB(const __bf16* base, int ld, int lane) {
  const int off = (lane & 15) * ld + ((lane >> 4) << 4);
  FragU f;
  f.u[0] = *(const v4u*)(base + off);
  f.u[1] = *(const v4u*)(base + off + 8);
  return f.v;
}

__device__ __forceinline__ v8f wmma_bf16(v16bf a, v16bf b, v8f c) {
  return __builtin_amdgcn_wmma_f32_16x16x32_bf16(false, a, false, b, (short)0, c,
                                                 false, false);
}

__device__ __forceinline__ v8f v8f_zero() { v8f z = {}; return z; }

// 16-byte async copy global -> LDS (tracked by ASYNCcnt).
// GVS addressing: mem = saddr + voffset; LDS dest = lds_addr VGPR.
__device__ __forceinline__ void async_copy16(unsigned lds_addr, unsigned voff,
                                             unsigned long long base) {
  asm volatile("global_load_async_to_lds_b128 %0, %1, %2 offset:0"
               :: "v"(lds_addr), "v"(voff), "s"(base) : "memory");
}

// ---------------------------------------------------------------- f32 -> bf16
__global__ void cvt_f32_bf16(const float* __restrict__ in, __bf16* __restrict__ out,
                             int n) {
  int i = blockIdx.x * 256 + threadIdx.x;
  if (i < n) out[i] = (__bf16)in[i];
}

// ---------------------------------------------------------------- GEMM
// C[m,n] = sum_k A[m,k] * W[n,k] + bias[n]
// MODE 0: bf16 row-major   MODE 1: sigmoid->bf16   MODE 2: f32 row-major
// MODE 3: bf16 store transposed into (B,H,DV,S)  (v for retention)
template <int MODE>
__global__ __launch_bounds__(256) void gemm_bf16_kernel(
    const __bf16* __restrict__ A, const __bf16* __restrict__ W,
    const float* __restrict__ bias, void* __restrict__ out,
    int Mdim, int Ndim, int Kdim, int nbn) {
  const int lane = threadIdx.x & 31;
  const int wid  = threadIdx.x >> 5;
  const int bm   = (int)blockIdx.x / nbn;
  const int bn   = (int)blockIdx.x % nbn;
  const int m0   = bm * 128 + wid * 16;
  const int n0   = bn * 64;

  const __bf16* Abase = A + (size_t)m0 * Kdim;
  const __bf16* W0 = W + (size_t)(n0 +  0) * Kdim;
  const __bf16* W1 = W + (size_t)(n0 + 16) * Kdim;
  const __bf16* W2 = W + (size_t)(n0 + 32) * Kdim;
  const __bf16* W3 = W + (size_t)(n0 + 48) * Kdim;

  v8f c0 = v8f_zero(), c1 = v8f_zero(), c2 = v8f_zero(), c3 = v8f_zero();

  v16bf a_c  = load_fragA(Abase, Kdim, lane);
  v16bf b0_c = load_fragB(W0, Kdim, lane);
  v16bf b1_c = load_fragB(W1, Kdim, lane);
  v16bf b2_c = load_fragB(W2, Kdim, lane);
  v16bf b3_c = load_fragB(W3, Kdim, lane);

  for (int k0 = 0; k0 < Kdim; k0 += 32) {
    const int k1 = (k0 + 32 < Kdim) ? k0 + 32 : k0;  // clamp: last iter reloads
    if (k0 + 256 < Kdim) {            // global_prefetch_b8 a few steps ahead
      __builtin_prefetch(Abase + k0 + 256, 0, 1);
      __builtin_prefetch(W0 + k0 + 256, 0, 1);
      __builtin_prefetch(W1 + k0 + 256, 0, 1);
      __builtin_prefetch(W2 + k0 + 256, 0, 1);
      __builtin_prefetch(W3 + k0 + 256, 0, 1);
    }
    v16bf a_n  = load_fragA(Abase + k1, Kdim, lane);
    v16bf b0_n = load_fragB(W0 + k1, Kdim, lane);
    v16bf b1_n = load_fragB(W1 + k1, Kdim, lane);
    v16bf b2_n = load_fragB(W2 + k1, Kdim, lane);
    v16bf b3_n = load_fragB(W3 + k1, Kdim, lane);
    c0 = wmma_bf16(a_c, b0_c, c0);
    c1 = wmma_bf16(a_c, b1_c, c1);
    c2 = wmma_bf16(a_c, b2_c, c2);
    c3 = wmma_bf16(a_c, b3_c, c3);
    a_c = a_n; b0_c = b0_n; b1_c = b1_n; b2_c = b2_n; b3_c = b3_n;
  }

  v8f cc[4] = {c0, c1, c2, c3};
  const int rbase = m0 + ((lane >> 4) << 3);
#pragma unroll
  for (int j = 0; j < 4; ++j) {
    const int col = n0 + 16 * j + (lane & 15);
    const float bv = bias[col];
#pragma unroll
    for (int r = 0; r < 8; ++r) {
      const int row = rbase + r;
      float v = cc[j][r] + bv;
      if constexpr (MODE == 0) {
        ((__bf16*)out)[(size_t)row * Ndim + col] = (__bf16)v;
      } else if constexpr (MODE == 1) {
        ((__bf16*)out)[(size_t)row * Ndim + col] =
            (__bf16)(1.0f / (1.0f + __expf(-v)));
      } else if constexpr (MODE == 2) {
        ((float*)out)[(size_t)row * Ndim + col] = v;
      } else {  // MODE 3: transpose-store v into (B,H,DV,S)
        const int b  = row / S, s  = row % S;
        const int h  = col / DV, dv = col % DV;
        ((__bf16*)out)[(size_t)((b * H + h) * DV + dv) * S + s] = (__bf16)v;
      }
    }
  }
}

// ---------------------------------------------------------------- RoPE (in place)
__global__ void rope_kernel(__bf16* __restrict__ buf) {
  const int total = B * S * H * (DK / 2);
  int idx = blockIdx.x * 256 + threadIdx.x;
  if (idx >= total) return;
  const int i = idx & (DK / 2 - 1);          // 0..63
  int rest    = idx >> 6;
  const int h = rest & (H - 1);
  rest >>= 3;
  const int s = rest & (S - 1);
  const int b = rest >> 11;
  const float freq = __expf(-(2.0f * i / (float)DK) * 9.210340371976184f); // ln 1e4
  const float ang  = (float)s * freq;
  const float cs = cosf(ang), sn = sinf(ang);
  __bf16* p = buf + ((size_t)(b * S + s) * E + h * DK + 2 * i);
  const float x1 = (float)p[0], x2 = (float)p[1];
  p[0] = (__bf16)(x1 * cs - x2 * sn);
  p[1] = (__bf16)(x1 * sn + x2 * cs);
}

// ---------------------------------------------------------------- retention core
// One wave per (b, h, 16-row query tile). Flash-style sweep over key tiles of 32.
// v tiles are staged into double-buffered LDS with async global->LDS copies a full
// iteration ahead (ASYNCcnt pipeline); fragments then come from ds_load_b128.
__global__ __launch_bounds__(32) void retention_kernel(
    const __bf16* __restrict__ q, const __bf16* __restrict__ k,
    const __bf16* __restrict__ vT, float* __restrict__ y) {
  const int lane = threadIdx.x;
  const int task = blockIdx.x;
  const int nt = task & (S / 16 - 1);       // 0..127
  const int hh = (task >> 7) & (H - 1);
  const int bb = task >> 10;
  const int n0 = nt * 16;

  const float gamma = 1.0f - exp2f(-5.0f - (float)hh);
  const float lg    = __logf(gamma);
  float gpow[8];
  {
    float gp = 1.0f;
#pragma unroll
    for (int r = 0; r < 8; ++r) { gpow[r] = gp; gp *= gamma; }
  }

  __shared__ alignas(16) __bf16 sc[16 * 32];         // score re-layout buffer
  __shared__ alignas(16) __bf16 vstage[2][32 * DV];  // [buf][dv*32 + key], 16KB ea

  // q A-fragments: 16 rows x DK=128 -> four 16x32 fragments
  v16bf aq[4];
#pragma unroll
  for (int kk = 0; kk < 4; ++kk)
    aq[kk] = load_fragA(q + ((size_t)bb * S + n0) * E + hh * DK + kk * 32, E, lane);

  v8f acc[16];
#pragma unroll
  for (int j = 0; j < 16; ++j) acc[j] = v8f_zero();

  const int rh = (lane >> 4) << 3;  // 0 or 8
  const int cn = lane & 15;
  const __bf16* kh  = k + (size_t)bb * S * E + hh * DK;
  const __bf16* vbh = vT + ((size_t)(bb * H + hh) * DV) * S;

  // ---- async staging setup: copy a 32(key) x 256(dv) tile (16KB) per iteration.
  // chunk c = lane + 32*t moves 16B: dv = c>>2, sub = c&3 (8 bf16 along key).
  const unsigned long long vbase = (unsigned long long)(uintptr_t)(const void*)vbh;
  const unsigned lds_v[2] = {
      (unsigned)(uintptr_t)(void*)&vstage[0][0],
      (unsigned)(uintptr_t)(void*)&vstage[1][0]};
  const unsigned voff_lane = (unsigned)((lane >> 2) * S * 2 + (lane & 3) * 16);
  const unsigned lds_lane  = (unsigned)((lane >> 2) * 64 + (lane & 3) * 16);

  auto stage_tile = [&](int m0t, unsigned lds_base) {
    const unsigned gb = voff_lane + (unsigned)(m0t * 2);
#pragma unroll
    for (int t = 0; t < 32; ++t)
      async_copy16(lds_base + lds_lane + t * 512, gb + t * (8 * S * 2), vbase);
  };

  stage_tile(0, lds_v[0]);
  int cur = 0;

  for (int m0 = 0; m0 < n0 + 16; m0 += 32) {
    asm volatile("s_wait_asynccnt 0x0" ::: "memory");  // buf[cur] ready
    // prefetch next tile into the other buffer (redundant clamp on last iter)
    const int m1 = (m0 + 32 < n0 + 16) ? m0 + 32 : m0;
    stage_tile(m1, lds_v[cur ^ 1]);

#pragma unroll
    for (int t = 0; t < 2; ++t) {
      const __bf16* kbase = kh + (size_t)(m0 + 16 * t) * E;
      v8f sacc = v8f_zero();
      v16bf bk_c = load_fragB(kbase, E, lane);
#pragma unroll
      for (int kk = 0; kk < 4; ++kk) {
        const int kn = (kk < 3) ? (kk + 1) * 32 : kk * 32;
        v16bf bk_n = load_fragB(kbase + kn, E, lane);
        sacc = wmma_bf16(aq[kk], bk_c, sacc);
        bk_c = bk_n;
      }
      // causal decay: gamma^(n-m) = gamma^(n0+rh-mc) * gamma^r (1 exp per tile)
      const int mc = m0 + 16 * t + cn;
      const float dbase = __expf((float)(n0 + rh - mc) * lg);
#pragma unroll
      for (int r = 0; r < 8; ++r) {
        const float d = (n0 + rh + r >= mc) ? dbase * gpow[r] : 0.0f;
        sc[(rh + r) * 32 + 16 * t + cn] = (__bf16)(sacc[r] * d);
      }
    }
    asm volatile("s_wait_dscnt 0x0" ::: "memory");   // sc visible to all lanes
    v16bf sa = load_fragA(sc, 32, lane);             // scores as bf16 A-fragment
    asm volatile("" ::: "memory");

    // v WMMAs from the staged LDS tile: B-frag = two ds_load_b128
    const __bf16* vt = &vstage[cur][0];
    v16bf bv_c = load_fragB(vt, 32, lane);           // tile rows: 32 keys, ld=32
#pragma unroll
    for (int j = 0; j < 16; ++j) {
      const int jn = (j < 15) ? j + 1 : j;
      v16bf bv_n = load_fragB(vt + (16 * jn) * 32, 32, lane);
      acc[j] = wmma_bf16(sa, bv_c, acc[j]);
      bv_c = bv_n;
    }
    cur ^= 1;
  }

  float* yo = y + ((size_t)bb * S + n0) * VE + hh * DV;
#pragma unroll
  for (int j = 0; j < 16; ++j)
#pragma unroll
    for (int r = 0; r < 8; ++r)
      yo[(size_t)(rh + r) * VE + 16 * j + cn] = acc[j][r];
}

// ------------------------------------------------- RMS norm * gn_weight * gate
__global__ __launch_bounds__(256) void rmsgate_kernel(
    const float* __restrict__ y, const __bf16* __restrict__ g,
    const float* __restrict__ gnw, __bf16* __restrict__ y2) {
  const int bs   = blockIdx.x;          // (b*S + s)
  const int h    = threadIdx.x >> 5;
  const int lane = threadIdx.x & 31;
  const float* yp = y + (size_t)bs * VE + h * DV;
  float vals[8];
  float ss = 0.0f;
#pragma unroll
  for (int i = 0; i < 8; ++i) {
    vals[i] = yp[lane + 32 * i];
    ss += vals[i] * vals[i];
  }
#pragma unroll
  for (int off = 16; off > 0; off >>= 1) ss += __shfl_xor(ss, off, 32);
  const float rms = rsqrtf(ss / (float)DV + 1e-6f);
#pragma unroll
  for (int i = 0; i < 8; ++i) {
    const int dv = lane + 32 * i;
    const size_t idx = (size_t)bs * VE + h * DV + dv;
    y2[idx] = (__bf16)(vals[i] * rms * gnw[dv] * (float)g[idx]);
  }
}

// ---------------------------------------------------------------- launcher
extern "C" void kernel_launch(void* const* d_in, const int* in_sizes, int n_in,
                              void* d_out, int out_size, void* d_ws, size_t ws_size,
                              hipStream_t stream) {
  (void)in_sizes; (void)n_in; (void)out_size; (void)ws_size;
  const float* x   = (const float*)d_in[0];
  const float* Wq  = (const float*)d_in[1];
  const float* bq  = (const float*)d_in[2];
  const float* Wk  = (const float*)d_in[3];
  const float* bk  = (const float*)d_in[4];
  const float* Wv  = (const float*)d_in[5];
  const float* bv  = (const float*)d_in[6];
  const float* Wg  = (const float*)d_in[7];
  const float* bg  = (const float*)d_in[8];
  const float* Wo  = (const float*)d_in[9];
  const float* bo  = (const float*)d_in[10];
  const float* gnw = (const float*)d_in[11];

  char* ws = (char*)d_ws;
  size_t off = 0;
  auto alloc = [&](size_t bytes) -> void* {
    void* p = ws + off;
    off += (bytes + 255) & ~(size_t)255;
    return p;
  };
  __bf16* xb = (__bf16*)alloc((size_t)M * E * 2);
  __bf16* wq = (__bf16*)alloc((size_t)E * E * 2);
  __bf16* wk = (__bf16*)alloc((size_t)E * E * 2);
  __bf16* wv = (__bf16*)alloc((size_t)VE * E * 2);
  __bf16* wg = (__bf16*)alloc((size_t)VE * E * 2);
  __bf16* wo = (__bf16*)alloc((size_t)E * VE * 2);
  __bf16* qb = (__bf16*)alloc((size_t)M * E * 2);
  __bf16* kb = (__bf16*)alloc((size_t)M * E * 2);   // contiguous after qb
  __bf16* vT = (__bf16*)alloc((size_t)B * VE * S * 2);
  __bf16* gb = (__bf16*)alloc((size_t)M * VE * 2);
  float*  yb = (float*)alloc((size_t)M * VE * 4);
  __bf16* y2 = qb;  // reuse q+k region (16 MB) after retention; q/k dead by then

  // 1. f32 -> bf16 conversions
  int n;
  n = M * E;  cvt_f32_bf16<<<(n + 255) / 256, 256, 0, stream>>>(x, xb, n);
  n = E * E;  cvt_f32_bf16<<<(n + 255) / 256, 256, 0, stream>>>(Wq, wq, n);
  n = E * E;  cvt_f32_bf16<<<(n + 255) / 256, 256, 0, stream>>>(Wk, wk, n);
  n = VE * E; cvt_f32_bf16<<<(n + 255) / 256, 256, 0, stream>>>(Wv, wv, n);
  n = VE * E; cvt_f32_bf16<<<(n + 255) / 256, 256, 0, stream>>>(Wg, wg, n);
  n = E * VE; cvt_f32_bf16<<<(n + 255) / 256, 256, 0, stream>>>(Wo, wo, n);

  // 2. projections (WMMA GEMMs)
  gemm_bf16_kernel<0><<<(M / 128) * (E / 64), 256, 0, stream>>>(
      xb, wq, bq, qb, M, E, E, E / 64);
  gemm_bf16_kernel<0><<<(M / 128) * (E / 64), 256, 0, stream>>>(
      xb, wk, bk, kb, M, E, E, E / 64);
  gemm_bf16_kernel<3><<<(M / 128) * (VE / 64), 256, 0, stream>>>(
      xb, wv, bv, vT, M, VE, E, VE / 64);
  gemm_bf16_kernel<1><<<(M / 128) * (VE / 64), 256, 0, stream>>>(
      xb, wg, bg, gb, M, VE, E, VE / 64);

  // 3. RoPE on q, k
  n = B * S * H * (DK / 2);
  rope_kernel<<<(n + 255) / 256, 256, 0, stream>>>(qb);
  rope_kernel<<<(n + 255) / 256, 256, 0, stream>>>(kb);

  // 4. retention
  retention_kernel<<<B * H * (S / 16), 32, 0, stream>>>(qb, kb, vT, yb);

  // 5. RMS-norm + gate
  rmsgate_kernel<<<M, 256, 0, stream>>>(yb, gb, gnw, y2);

  // 6. output projection -> d_out (f32)
  gemm_bf16_kernel<2><<<(M / 128) * (E / 64), 256, 0, stream>>>(
      y2, wo, bo, d_out, M, E, VE, E / 64);
}
